// MoE_64536178590140
// MI455X (gfx1250) — matrix operand order repaired and checked by reading, not compiled, not run on hardware
//
#include <hip/hip_runtime.h>

// ---------------- problem constants ----------------
#define BSZ   4
#define NSEQ  1024
#define DIM   1024
#define NEXP  8
#define TOPN  2
#define CAP   256
#define MTOT  (BSZ * CAP)          // 1024 rows per expert
#define HH    2730
#define HH2   (2 * HH)             // 5460
#define HPAD  2752                 // 43*64, padded GEGLU width (mult of 64)
#define BN    (BSZ * NSEQ)         // 4096 tokens

typedef __attribute__((ext_vector_type(16))) __bf16 v16bf;
typedef __attribute__((ext_vector_type(8)))  float  v8f;

union FragU { unsigned int u[8]; v16bf v; };

__device__ __forceinline__ unsigned int f2bf(float f) {
    unsigned int u = __float_as_uint(f);
    return (u + 0x7FFFu + ((u >> 16) & 1u)) >> 16;   // RNE
}

// ---- CDNA5 async global->LDS copy (ASYNCcnt path), guarded fallback ----
#if __has_builtin(__builtin_amdgcn_global_load_async_to_lds_b32) && \
    __has_builtin(__builtin_amdgcn_s_wait_asynccnt)
#define HAVE_ASYNC_LDS 1
#define AS1 __attribute__((address_space(1)))
#define AS3 __attribute__((address_space(3)))
#endif

__device__ __forceinline__ void g2l_b32(unsigned int* lds, const unsigned int* g) {
#ifdef HAVE_ASYNC_LDS
    __builtin_amdgcn_global_load_async_to_lds_b32(
        (AS1 int*)(uintptr_t)g,
        (AS3 int*)(unsigned int)(uintptr_t)lds,
        0, 0);
#else
    *lds = *g;
#endif
}

__device__ __forceinline__ void async_wait() {
#ifdef HAVE_ASYNC_LDS
    __builtin_amdgcn_s_wait_asynccnt(0);
#endif
}

// ---------------- 1) gating: logits -> softmax -> top2 -> route flags ----------------
__global__ __launch_bounds__(256) void k_gating(
    const float* __restrict__ x,      // [B,N,D]
    const float* __restrict__ rp,     // [TOPN,B,N]
    const float* __restrict__ gw,     // [D,E]
    int* __restrict__ tok_exp,        // [TOPN,BN]
    int* __restrict__ tok_rt,         // [TOPN,BN]
    float* __restrict__ tok_g,        // [TOPN,BN]
    int* __restrict__ tok_slot)       // [TOPN,BN]
{
    int t = blockIdx.x * blockDim.x + threadIdx.x;
    if (t >= BN) return;

    float lg[NEXP];
#pragma unroll
    for (int e = 0; e < NEXP; ++e) lg[e] = 0.f;
    const float* xr = x + (size_t)t * DIM;
    for (int d = 0; d < DIM; ++d) {
        float xv = xr[d];
        const float* g = gw + (size_t)d * NEXP;
#pragma unroll
        for (int e = 0; e < NEXP; ++e) lg[e] = fmaf(xv, g[e], lg[e]);
    }
    float mx = lg[0];
#pragma unroll
    for (int e = 1; e < NEXP; ++e) mx = fmaxf(mx, lg[e]);
    float p[NEXP], s = 0.f;
#pragma unroll
    for (int e = 0; e < NEXP; ++e) { p[e] = __expf(lg[e] - mx); s += p[e]; }

    int e0 = 0; float g0 = p[0];
#pragma unroll
    for (int e = 1; e < NEXP; ++e) if (p[e] > g0) { g0 = p[e]; e0 = e; }
    int e1 = -1; float g1 = -1.f;
#pragma unroll
    for (int e = 0; e < NEXP; ++e) if (e != e0 && p[e] > g1) { g1 = p[e]; e1 = e; }

    g0 /= s; g1 /= s;
    float denom = fmaxf(g0 + g1, 1e-9f);
    float gv0 = g0 / denom, gv1 = g1 / denom;

    int rt1 = (rp[(size_t)1 * BN + t] < (gv1 / 0.2f)) ? 1 : 0;

    tok_exp[t] = e0;       tok_exp[BN + t] = e1;
    tok_g[t]   = gv0;      tok_g[BN + t]   = gv1;
    tok_rt[t]  = 1;        tok_rt[BN + t]  = rt1;
    tok_slot[t] = -1;      tok_slot[BN + t] = -1;
}

// ---------------- 2) capacity assignment (serial scan, 1 wave, (b,e) per lane) ----------------
__global__ void k_assign(
    const int* __restrict__ tok_exp,
    const int* __restrict__ tok_rt,
    int* __restrict__ tok_slot,       // [TOPN,BN]
    int* __restrict__ slot_token)     // [E,B,CAP] -> token n or -1
{
    int tid = threadIdx.x;
    if (tid >= BSZ * NEXP) return;
    int b = tid >> 3, e = tid & 7;
    int* st = slot_token + ((size_t)e * BSZ + b) * CAP;
    for (int c = 0; c < CAP; ++c) st[c] = -1;

    int prev = 0;
    for (int k = 0; k < TOPN; ++k) {
        int inpass = 0, kept = 0;
        for (int n = 0; n < NSEQ; ++n) {
            int idx = k * BN + b * NSEQ + n;
            if (tok_exp[idx] == e && tok_rt[idx]) {
                int pos = inpass + prev;   // exclusive cumsum incl. dropped + prev kept
                ++inpass;
                if (pos < CAP) { tok_slot[idx] = pos; st[pos] = n; ++kept; }
            }
        }
        prev += kept;
    }
}

// ---------------- 3) dispatch: gather tokens -> bf16 xe[E][MTOT][DIM] ----------------
__global__ __launch_bounds__(256) void k_dispatch(
    const float* __restrict__ x,
    const int* __restrict__ slot_token,
    unsigned int* __restrict__ xe)    // bf16 pairs: [E*MTOT][DIM/2]
{
    int row = blockIdx.x;                       // e*MTOT + b*CAP + c
    int e = row >> 10, m = row & 1023;
    int b = m >> 8,  c = m & 255;
    int tok = slot_token[((size_t)e * BSZ + b) * CAP + c];
    unsigned int* dst = xe + (size_t)row * (DIM / 2);
    int tid = threadIdx.x;
    if (tok < 0) {
        for (int i = tid; i < DIM / 2; i += 256) dst[i] = 0u;
    } else {
        const float* xr = x + ((size_t)b * NSEQ + tok) * DIM;
        for (int i = tid; i < DIM / 2; i += 256)
            dst[i] = f2bf(xr[2 * i]) | (f2bf(xr[2 * i + 1]) << 16);
    }
}

// ---------------- 4) GEMM1 + GEGLU (bf16 WMMA), act -> bf16 [E][MTOT][HPAD] ----------------
__global__ __launch_bounds__(256) void k_gemm1_geglu(
    const unsigned int* __restrict__ xe,    // bf16 pairs [E*MTOT][DIM/2]
    const float* __restrict__ w1,           // [E][DIM][HH2]
    const float* __restrict__ b1,           // [E][HH2]
    const float* __restrict__ mb,           // [E][HH]
    unsigned int* __restrict__ act)         // bf16 pairs [E*MTOT][HPAD/2]
{
    const int e     = blockIdx.z;
    const int Mbase = blockIdx.y * 64;
    const int Jbase = blockIdx.x * 64;      // logical col in [0,HPAD)
    const int tid   = threadIdx.x;
    const int wave  = tid >> 5, lane = tid & 31;
    const int hl    = lane >> 4, lr = lane & 15;
    const int strip = wave >> 2;            // 0 = val cols, 1 = gate cols (H + j)
    const int mrow  = (wave & 3) << 4;

    __shared__ unsigned int As[64][17];       // 64 rows x 16 K-pairs (+pad)
    __shared__ unsigned int Bs[2][64][17];    // [strip][col][K-pair], transposed
    __shared__ float Hs[2][64][64];           // h exchange (val/gate)

    v8f acc[4];
#pragma unroll
    for (int t = 0; t < 4; ++t) acc[t] = (v8f)0.f;

    const unsigned int* xrow = xe + (size_t)e * MTOT * (DIM / 2);
    const float* w1e = w1 + (size_t)e * DIM * HH2;

    for (int kb = 0; kb < DIM; kb += 32) {
#pragma unroll
        for (int i = 0; i < 4; ++i) {         // A tile 64x32 bf16 (async global->LDS)
            int idx = tid + i * 256;
            int r = idx >> 4, kp = idx & 15;
            g2l_b32(&As[r][kp],
                    &xrow[(size_t)(Mbase + r) * (DIM / 2) + (kb >> 1) + kp]);
        }
#pragma unroll
        for (int s = 0; s < 2; ++s) {         // W tiles 32x64, fp32 -> bf16, transposed
#pragma unroll
            for (int i = 0; i < 4; ++i) {
                int idx = tid + i * 256;
                int col = idx >> 4, kp = idx & 15;
                int j = Jbase + col;
                unsigned int pack = 0u;
                if (j < HH) {
                    int gcol = (s == 0) ? j : (HH + j);
                    float f0 = w1e[(size_t)(kb + 2 * kp)     * HH2 + gcol];
                    float f1 = w1e[(size_t)(kb + 2 * kp + 1) * HH2 + gcol];
                    pack = f2bf(f0) | (f2bf(f1) << 16);
                }
                Bs[s][col][kp] = pack;
            }
        }
        if (kb + 32 < DIM)
            __builtin_prefetch(&w1e[(size_t)(kb + 32) * HH2 + Jbase], 0, 0);
        async_wait();
        __syncthreads();

        FragU a;
#pragma unroll
        for (int v = 0; v < 8; ++v) {
            int kp = ((v & 4) << 1) + (hl << 2) + (v & 3);
            a.u[v] = As[mrow + lr][kp];
        }
#pragma unroll
        for (int t = 0; t < 4; ++t) {
            FragU bf;
#pragma unroll
            for (int v = 0; v < 8; ++v)
                bf.u[v] = Bs[strip][t * 16 + lr][((v & 4) << 1) + (hl << 2) + (v & 3)];
            acc[t] = __builtin_amdgcn_wmma_f32_16x16x32_bf16(
                false, a.v, false, bf.v, (short)0, acc[t], false, false);
        }
        __syncthreads();
    }

    // exchange h through LDS
#pragma unroll
    for (int t = 0; t < 4; ++t)
#pragma unroll
        for (int i = 0; i < 8; ++i)
            Hs[strip][mrow + hl * 8 + i][t * 16 + lr] = acc[t][i];
    __syncthreads();

    // GEGLU (exact gelu) + bf16 store; cols >= HH store zero (covers padding)
    const float* b1e = b1 + (size_t)e * HH2;
    const float* mbe = mb + (size_t)e * HH;
    for (int idx = tid; idx < 64 * 32; idx += 256) {
        int r = idx >> 5, cp = idx & 31;
        int j0 = Jbase + 2 * cp;
        unsigned int lo = 0u, hi = 0u;
#pragma unroll
        for (int h = 0; h < 2; ++h) {
            int j = j0 + h;
            if (j < HH) {
                float val = Hs[0][r][2 * cp + h] + b1e[j];
                float g   = Hs[1][r][2 * cp + h] + b1e[HH + j];
                float act_v = 0.5f * g * (1.0f + erff(g * 0.70710678118654752f))
                              * val * mbe[j];
                if (h == 0) lo = f2bf(act_v); else hi = f2bf(act_v);
            }
        }
        act[(size_t)(e * MTOT + Mbase + r) * (HPAD / 2) + (Jbase >> 1) + cp] =
            lo | (hi << 16);
    }
}

// ---------------- 5) GEMM2 (bf16 WMMA): out_e = act @ w2 + b2 ----------------
__global__ __launch_bounds__(256) void k_gemm2(
    const unsigned int* __restrict__ act,   // bf16 pairs [E*MTOT][HPAD/2]
    const float* __restrict__ w2,           // [E][HH][DIM]
    const float* __restrict__ b2,           // [E][DIM]
    float* __restrict__ out_e)              // [E][MTOT][DIM]
{
    const int e     = blockIdx.z;
    const int Mbase = blockIdx.y * 64;
    const int Cbase = blockIdx.x * 64;
    const int tid   = threadIdx.x;
    const int wave  = tid >> 5, lane = tid & 31;
    const int hl    = lane >> 4, lr = lane & 15;
    const int mrow  = (wave & 3) << 4;
    const int chalf = wave >> 2;            // 0/1 -> 32-col half

    __shared__ unsigned int As[64][17];
    __shared__ unsigned int Bs[64][17];

    v8f acc[2];
    acc[0] = (v8f)0.f; acc[1] = (v8f)0.f;

    const unsigned int* arow = act + (size_t)e * MTOT * (HPAD / 2);
    const float* w2e = w2 + (size_t)e * HH * DIM;

    for (int kb = 0; kb < HPAD; kb += 32) {
#pragma unroll
        for (int i = 0; i < 4; ++i) {        // A tile (async global->LDS)
            int idx = tid + i * 256;
            int r = idx >> 4, kp = idx & 15;
            g2l_b32(&As[r][kp],
                    &arow[(size_t)(Mbase + r) * (HPAD / 2) + (kb >> 1) + kp]);
        }
#pragma unroll
        for (int i = 0; i < 4; ++i) {
            int idx = tid + i * 256;
            int col = idx >> 4, kp = idx & 15;
            int k0 = kb + 2 * kp;
            float f0 = (k0     < HH) ? w2e[(size_t)k0       * DIM + Cbase + col] : 0.f;
            float f1 = (k0 + 1 < HH) ? w2e[(size_t)(k0 + 1) * DIM + Cbase + col] : 0.f;
            Bs[col][kp] = f2bf(f0) | (f2bf(f1) << 16);
        }
        if (kb + 32 < HPAD && kb + 32 < HH)
            __builtin_prefetch(&w2e[(size_t)(kb + 32) * DIM + Cbase], 0, 0);
        async_wait();
        __syncthreads();

        FragU a;
#pragma unroll
        for (int v = 0; v < 8; ++v) {
            int kp = ((v & 4) << 1) + (hl << 2) + (v & 3);
            a.u[v] = As[mrow + lr][kp];
        }
#pragma unroll
        for (int t = 0; t < 2; ++t) {
            FragU bf;
#pragma unroll
            for (int v = 0; v < 8; ++v)
                bf.u[v] = Bs[chalf * 32 + t * 16 + lr][((v & 4) << 1) + (hl << 2) + (v & 3)];
            acc[t] = __builtin_amdgcn_wmma_f32_16x16x32_bf16(
                false, a.v, false, bf.v, (short)0, acc[t], false, false);
        }
        __syncthreads();
    }

    const float* b2e = b2 + (size_t)e * DIM;
#pragma unroll
    for (int t = 0; t < 2; ++t) {
        int col = Cbase + chalf * 32 + t * 16 + lr;
#pragma unroll
        for (int i = 0; i < 8; ++i) {
            int row = Mbase + mrow + hl * 8 + i;
            out_e[((size_t)e * MTOT + row) * DIM + col] = acc[t][i] + b2e[col];
        }
    }
}

// ---------------- 6) combine: weighted gather back to tokens ----------------
__global__ __launch_bounds__(256) void k_combine(
    const float* __restrict__ out_e,
    const int* __restrict__ tok_exp,
    const int* __restrict__ tok_slot,
    const float* __restrict__ tok_g,
    float* __restrict__ out)          // [B,N,D]
{
    int t = blockIdx.x;               // b*N + n
    int b = t >> 10;
    int d0 = threadIdx.x * 4;
    float acc[4] = {0.f, 0.f, 0.f, 0.f};
#pragma unroll
    for (int k = 0; k < TOPN; ++k) {
        int idx = k * BN + t;
        int slot = tok_slot[idx];
        if (slot >= 0) {
            int e = tok_exp[idx];
            float g = tok_g[idx];
            const float* src = out_e +
                ((size_t)e * MTOT + (size_t)b * CAP + slot) * DIM + d0;
#pragma unroll
            for (int i = 0; i < 4; ++i) acc[i] = fmaf(g, src[i], acc[i]);
        }
    }
    float* dst = out + (size_t)t * DIM + d0;
#pragma unroll
    for (int i = 0; i < 4; ++i) dst[i] = acc[i];
}

// ---------------- host-side launch ----------------
extern "C" void kernel_launch(void* const* d_in, const int* in_sizes, int n_in,
                              void* d_out, int out_size, void* d_ws, size_t ws_size,
                              hipStream_t stream) {
    (void)in_sizes; (void)n_in; (void)out_size; (void)ws_size;

    const float* x   = (const float*)d_in[0];
    const float* rp  = (const float*)d_in[1];
    const float* gw  = (const float*)d_in[2];
    const float* w1  = (const float*)d_in[3];
    const float* b1  = (const float*)d_in[4];
    const float* mb  = (const float*)d_in[5];
    const float* w2  = (const float*)d_in[6];
    const float* b2  = (const float*)d_in[7];
    float* out = (float*)d_out;

    char* ws = (char*)d_ws;
    size_t off = 0;
    auto carve = [&](size_t bytes) -> void* {
        void* p = ws + off;
        off = (off + bytes + 255) & ~(size_t)255;
        return p;
    };
    int*   tok_exp    = (int*)  carve(sizeof(int)   * TOPN * BN);
    int*   tok_rt     = (int*)  carve(sizeof(int)   * TOPN * BN);
    int*   tok_slot   = (int*)  carve(sizeof(int)   * TOPN * BN);
    float* tok_g      = (float*)carve(sizeof(float) * TOPN * BN);
    int*   slot_token = (int*)  carve(sizeof(int)   * NEXP * BSZ * CAP);
    unsigned int* xe  = (unsigned int*)carve(sizeof(unsigned int) * (size_t)NEXP * MTOT * (DIM / 2));
    unsigned int* act = (unsigned int*)carve(sizeof(unsigned int) * (size_t)NEXP * MTOT * (HPAD / 2));
    float* out_e      = (float*)carve(sizeof(float) * (size_t)NEXP * MTOT * DIM);

    k_gating  <<<BN / 256, 256, 0, stream>>>(x, rp, gw, tok_exp, tok_rt, tok_g, tok_slot);
    k_assign  <<<1, 32, 0, stream>>>(tok_exp, tok_rt, tok_slot, slot_token);
    k_dispatch<<<NEXP * MTOT, 256, 0, stream>>>(x, slot_token, xe);
    k_gemm1_geglu<<<dim3(HPAD / 64, MTOT / 64, NEXP), 256, 0, stream>>>(xe, w1, b1, mb, act);
    k_gemm2   <<<dim3(DIM / 64, MTOT / 64, NEXP), 256, 0, stream>>>(act, w2, b2, out_e);
    k_combine <<<BN, 256, 0, stream>>>(out_e, tok_exp, tok_slot, tok_g, out);
}
